// MAtt_22643067584702
// MI455X (gfx1250) — matrix operand order repaired and checked by reading, not compiled, not run on hardware
//
#include <hip/hip_runtime.h>
#include <math.h>
#include <stdint.h>

typedef __attribute__((ext_vector_type(16))) _Float16     v16h;
typedef __attribute__((ext_vector_type(8)))  float        v8f;
typedef __attribute__((ext_vector_type(4)))  unsigned int u32x4;
typedef __attribute__((ext_vector_type(8)))  int          i32x8;
typedef __attribute__((ext_vector_type(4)))  int          i32x4;

// ---------------- problem constants ----------------
#define BATCH   1024
#define C_IN    22
#define T_IN    2047
#define C_TEMP  20
#define KT      12
#define T_OUT   2048          // 2047 + 2*6 - 12 + 1
#define N_PATCH 8
#define TP      256           // T_OUT / N_PATCH
#define ATT_N   18
#define KPACK   264           // C_IN*KT
#define KPAD    288           // 9 * 32
#define MPAD    32
#define N_OUT   4
#define TANG    171           // 18*19/2
#define FEATD   1368          // 8*171

// ---------------- stats layout (floats, in ws[0..1023]) ----------------
#define SX    0     // 22  sum x per channel
#define GOFF  32    // 484 upper-tri gram (i<=j at i*22+j)
#define SC1   520   // 22  BN1 scale
#define SH1   544   // 22  BN1 fused shift
#define SY    576   // 20  sum conv-out
#define SY2   608   // 20  sum conv-out^2
#define SC2   640   // 20  BN2 scale

// ---------------- ws byte offsets ----------------
#define OFF_STAT  ((size_t)0)
#define OFF_AW    ((size_t)4096)                         // 32*288 f16 = 18432
#define OFF_H1    ((size_t)22528)                        // 1024*22*2047 f16 = 92,229,632
#define OFF_Y     ((size_t)92241920)                     // 1024*20*2048 f32 = 167,772,160
#define OFF_COV   ((size_t)260014080)                    // 1024*8*400 f32  = 13,107,200
#define OFF_LQ    ((size_t)273121280)                    // 1024*8*324 f32  = 10,616,832
#define OFF_LK    ((size_t)283738112)
#define OFF_LV    ((size_t)294354944)
#define OFF_FEAT  ((size_t)304971776)                    // 1024*1368 f32   = 5,603,328

__device__ __forceinline__ void tri_decode(int e, int n, int& i, int& j) {
    int ii = 0, off = 0;
    while (e >= off + (n - ii)) { off += (n - ii); ++ii; }
    i = ii; j = ii + (e - off);
}

// =====================================================================
// K_w: pack temporal-conv weights into padded 32x288 f16 GEMM-A matrix
// =====================================================================
__global__ void prep_weights(const float* __restrict__ w_temp, _Float16* __restrict__ Aw) {
    int e = blockIdx.x * 256 + threadIdx.x;        // < 32*288
    if (e >= MPAD * KPAD) return;
    int o = e / KPAD, k = e - o * KPAD;
    float v = (o < C_TEMP && k < KPACK) ? w_temp[o * KPACK + k] : 0.f;
    Aw[e] = (_Float16)v;
}

// =====================================================================
// K0: channel sums + 22x22 Gram of x  (exact BN1 stats via linearity)
// =====================================================================
__global__ void gram_x(const float* __restrict__ x, float* __restrict__ st) {
    __shared__ float xs[C_IN * 256];
    int b = blockIdx.x >> 3, ch = blockIdx.x & 7;
    int t0 = ch * 256, tid = threadIdx.x;
    for (int e = tid; e < C_IN * 256; e += 256) {
        int c = e >> 8, t = e & 255, gt = t0 + t;
        xs[e] = (gt < T_IN) ? x[((size_t)b * C_IN + c) * T_IN + gt] : 0.f;
    }
    __syncthreads();
    if (tid < C_IN) {
        float s = 0.f;
        for (int t = 0; t < 256; ++t) s += xs[tid * 256 + t];
        atomicAdd(&st[SX + tid], s);
    } else if (tid < C_IN + 253) {
        int i, j; tri_decode(tid - C_IN, C_IN, i, j);
        float s = 0.f;
        for (int t = 0; t < 256; ++t) s += xs[i * 256 + t] * xs[j * 256 + t];
        atomicAdd(&st[GOFF + i * C_IN + j], s);
    }
}

// =====================================================================
// K0b: finalize BN1 scale/shift (folded with spatial bias)
// =====================================================================
__global__ void bn1_finalize(const float* __restrict__ w_spat, const float* __restrict__ b_spat,
                             const float* __restrict__ g1, const float* __restrict__ be1,
                             float* __restrict__ st) {
    int o = threadIdx.x;
    if (o >= C_IN) return;
    const float invN = 1.f / ((float)BATCH * (float)T_IN);
    float wm = 0.f;
    for (int c = 0; c < C_IN; ++c) wm += w_spat[o * C_IN + c] * (st[SX + c] * invN);
    float b  = b_spat[o];
    float mu = wm + b;
    float q = 0.f;
    for (int c1 = 0; c1 < C_IN; ++c1)
        for (int c2 = 0; c2 < C_IN; ++c2) {
            int a  = (c1 <= c2) ? c1 : c2;
            int bb = (c1 <= c2) ? c2 : c1;
            q += w_spat[o * C_IN + c1] * w_spat[o * C_IN + c2] * st[GOFF + a * C_IN + bb];
        }
    float Eh2 = q * invN + 2.f * b * wm + b * b;
    float var = Eh2 - mu * mu;
    float sc  = g1[o] * rsqrtf(var + 1e-5f);
    st[SC1 + o] = sc;
    st[SH1 + o] = (b - mu) * sc + be1[o];
}

// =====================================================================
// K1: fused spatial filter + BN1 -> f16 activations
// =====================================================================
__global__ void spatial_bn(const float* __restrict__ x, const float* __restrict__ w_spat,
                           const float* __restrict__ st, _Float16* __restrict__ h1) {
    __shared__ float xs[C_IN * 256];
    __shared__ float wl[C_IN * C_IN];
    __shared__ float scl[C_IN], shl[C_IN];
    int b = blockIdx.x >> 3, ch = blockIdx.x & 7;
    int t0 = ch * 256, tid = threadIdx.x;
    for (int e = tid; e < C_IN * C_IN; e += 256) wl[e] = w_spat[e];
    if (tid < C_IN) { scl[tid] = st[SC1 + tid]; shl[tid] = st[SH1 + tid]; }
    for (int e = tid; e < C_IN * 256; e += 256) {
        int c = e >> 8, t = e & 255, gt = t0 + t;
        xs[e] = (gt < T_IN) ? x[((size_t)b * C_IN + c) * T_IN + gt] : 0.f;
    }
    __syncthreads();
    int gt = t0 + tid;
    if (gt < T_IN) {
        for (int o = 0; o < C_IN; ++o) {
            float s = 0.f;
            for (int c = 0; c < C_IN; ++c) s += wl[o * C_IN + c] * xs[c * 256 + tid];
            h1[((size_t)b * C_IN + o) * T_IN + gt] = (_Float16)(s * scl[o] + shl[o]);
        }
    }
}

// =====================================================================
// K2: temporal conv as WMMA GEMM  (20x264) x (264 x 128-time tile)
//     weights staged into LDS by the Tensor Data Mover (one 4x288 f16
//     slice per wave), activations by guarded VALU fill (zero padding)
// =====================================================================
#define NTILE 128
#define HS_W  144
__global__ void conv_wmma(const _Float16* __restrict__ Aw, const _Float16* __restrict__ h1,
                          float* __restrict__ y) {
    __shared__ __align__(16) _Float16 As[MPAD * KPAD];   // 18432 B
    __shared__ __align__(16) _Float16 Hs[C_IN * HS_W];   //  6336 B
    int b = blockIdx.x >> 4, tile = blockIdx.x & 15;
    int t0 = tile * NTILE;
    int tid = threadIdx.x;
    int wave = tid >> 5, lane = tid & 31;

    // ---- TDM: D# per ISA 8.3/8.4; wave w moves rows [4w, 4w+4) of A ----
    {
        unsigned long long ga = (unsigned long long)(uintptr_t)Aw
                              + (unsigned long long)wave * 2304ull;     // 4*288*2 B
        unsigned lds = (unsigned)(uintptr_t)As + (unsigned)wave * 2304u; // LDS byte addr
        u32x4 g0;
        g0[0] = 1u;                                   // count=1 (valid user D#)
        g0[1] = lds;                                  // lds_addr
        g0[2] = (unsigned)(ga & 0xffffffffull);       // global_addr[31:0]
        g0[3] = (unsigned)(ga >> 32) | 0x80000000u;   // global_addr[56:32] | type=2<<30
        i32x8 g1;
        g1[0] = 0x00010000;                           // data_size = 2 bytes
        g1[1] = (int)(288u << 16);                    // tensor_dim0 = 288 (elements)
        g1[2] = (int)(32u << 16);                     // tensor_dim0_hi=0 | tensor_dim1=32
        g1[3] = (int)(288u << 16);                    // tensor_dim1_hi=0 | tile_dim0=288
        g1[4] = 4;                                    // tile_dim1=4, tile_dim2=unused
        g1[5] = 288;                                  // tensor_dim0_stride (elements)
        g1[6] = 0;
        g1[7] = 0;
        i32x4 gz4 = {0, 0, 0, 0};                     // <=2D: groups 2/3 zero
        i32x8 gz8 = {0, 0, 0, 0, 0, 0, 0, 0};
        __builtin_amdgcn_tensor_load_to_lds(g0, g1, gz4, gz4, gz8, 0);
    }

    // ---- activation window [t0-6, t0+137] with zero padding ----
    for (int e = tid; e < C_IN * HS_W; e += 256) {
        int c = e / HS_W, j = e - c * HS_W;
        int gt = t0 - 6 + j;
        Hs[e] = (gt >= 0 && gt < T_IN) ? h1[((size_t)b * C_IN + c) * T_IN + gt] : (_Float16)0.f;
    }
    __builtin_amdgcn_s_wait_tensorcnt(0);
    __syncthreads();

    int n  = lane & 15;
    int hi = lane >> 4;
    v8f acc0 = {0.f,0.f,0.f,0.f,0.f,0.f,0.f,0.f};
    v8f acc1 = {0.f,0.f,0.f,0.f,0.f,0.f,0.f,0.f};
    int tl = wave * 16 + n;                  // local output time within tile
    const unsigned* As32 = (const unsigned*)As;

    for (int kt = 0; kt < 9; ++kt) {
        // A fragments: 16-bit A 16x32 layout, dword (half-pair) LDS loads
        union UH { v16h h; unsigned u[8]; };
        UH ua0, ua1;
        int m    = lane & 15;
        int kbA  = kt * 32 + hi * 8;
        int rb0  = (m * KPAD + kbA) >> 1;
        int rb1  = ((m + 16) * KPAD + kbA) >> 1;
        #pragma unroll
        for (int d = 0; d < 4; ++d) {
            ua0.u[d]     = As32[rb0 + d];
            ua0.u[d + 4] = As32[rb0 + 8 + d];
            ua1.u[d]     = As32[rb1 + d];
            ua1.u[d + 4] = As32[rb1 + 8 + d];
        }
        // B fragment: 16-bit B 32x16 layout, gathered from activation window
        v16h bf;
        int kbB = kt * 32 + hi * 8;
        #pragma unroll
        for (int jj = 0; jj < 16; ++jj) {
            int k = kbB + (jj < 8 ? jj : jj + 8);
            _Float16 v = (_Float16)0.f;
            if (k < KPACK) {
                int ci = k / KT;
                int kw = k - ci * KT;
                v = Hs[ci * HS_W + tl + kw]; // = h[ci][t0 + tl + kw - 6]
            }
            bf[jj] = v;
        }
        acc0 = __builtin_amdgcn_wmma_f32_16x16x32_f16(false, ua0.h, false, bf, (short)0, acc0, false, false);
        acc1 = __builtin_amdgcn_wmma_f32_16x16x32_f16(false, ua1.h, false, bf, (short)0, acc1, false, false);
    }

    int tcol = t0 + tl;
    #pragma unroll
    for (int r = 0; r < 8; ++r) {
        int m0 = r + hi * 8;                 // C/D layout: lanes16-31 hold M+8
        if (m0 < C_TEMP) y[((size_t)b * C_TEMP + m0) * T_OUT + tcol] = acc0[r];
        int m1 = 16 + r + hi * 8;
        if (m1 < C_TEMP) y[((size_t)b * C_TEMP + m1) * T_OUT + tcol] = acc1[r];
    }
}

// =====================================================================
// K3: BN2 variance reduction (mean/shift cancel under patch centering)
// =====================================================================
__global__ void bn2_stats(const float* __restrict__ y, float* __restrict__ st) {
    __shared__ float s1[256], s2[256];
    int bo = blockIdx.x;
    int b = bo / C_TEMP, o = bo - b * C_TEMP;
    const float* row = y + ((size_t)b * C_TEMP + o) * T_OUT;
    float a1 = 0.f, a2 = 0.f;
    for (int t = threadIdx.x; t < T_OUT; t += 256) { float v = row[t]; a1 += v; a2 += v * v; }
    s1[threadIdx.x] = a1; s2[threadIdx.x] = a2;
    __syncthreads();
    for (int s = 128; s > 0; s >>= 1) {
        if (threadIdx.x < s) { s1[threadIdx.x] += s1[threadIdx.x + s]; s2[threadIdx.x] += s2[threadIdx.x + s]; }
        __syncthreads();
    }
    if (threadIdx.x == 0) { atomicAdd(&st[SY + o], s1[0]); atomicAdd(&st[SY2 + o], s2[0]); }
}

__global__ void bn2_finalize(const float* __restrict__ g2, float* __restrict__ st) {
    int o = threadIdx.x;
    if (o >= C_TEMP) return;
    const float invN = 1.f / ((float)BATCH * (float)T_OUT);
    float mean = st[SY + o] * invN;
    float var  = st[SY2 + o] * invN - mean * mean;
    st[SC2 + o] = g2[o] * rsqrtf(var + 1e-5f);
}

// =====================================================================
// K4: per-(b,p) scaled, centered covariance + trace normalization
// =====================================================================
__global__ void cov_k(const float* __restrict__ y, const float* __restrict__ st,
                      float* __restrict__ cov) {
    __shared__ float ht[C_TEMP * TP];
    __shared__ float mn[C_TEMP];
    __shared__ float cv[C_TEMP * C_TEMP];
    __shared__ float invtr;
    int bp = blockIdx.x;
    int b = bp >> 3, p = bp & 7;
    int tid = threadIdx.x;
    for (int e = tid; e < C_TEMP * TP; e += 256) {
        int c = e >> 8, t = e & 255;
        ht[e] = y[((size_t)b * C_TEMP + c) * T_OUT + p * TP + t] * st[SC2 + c];
    }
    __syncthreads();
    if (tid < C_TEMP) {
        float s = 0.f;
        for (int t = 0; t < TP; ++t) s += ht[tid * TP + t];
        mn[tid] = s * (1.f / (float)TP);
    }
    __syncthreads();
    for (int e = tid; e < C_TEMP * TP; e += 256) ht[e] -= mn[e >> 8];
    __syncthreads();
    if (tid < 210) {
        int i, j; tri_decode(tid, C_TEMP, i, j);
        float s = 0.f;
        for (int t = 0; t < TP; ++t) s += ht[i * TP + t] * ht[j * TP + t];
        s *= (1.f / (float)(TP - 1));
        cv[i * C_TEMP + j] = s; cv[j * C_TEMP + i] = s;
    }
    __syncthreads();
    if (tid == 0) {
        float tr = 0.f;
        for (int c = 0; c < C_TEMP; ++c) tr += cv[c * C_TEMP + c];
        invtr = 1.f / tr;
    }
    __syncthreads();
    for (int e = tid; e < C_TEMP * C_TEMP; e += 256)
        cov[(size_t)bp * 400 + e] = cv[e] * invtr;
}

// =====================================================================
// wave32-parallel cyclic Jacobi eigensolver for 18x18 symmetric (LDS)
// =====================================================================
#define JLD 19
__device__ void jacobi18(float* A, float* V, int lane) {
    for (int e = lane; e < ATT_N * JLD; e += 32) V[e] = 0.f;
    __builtin_amdgcn_wave_barrier();
    if (lane < ATT_N) V[lane * JLD + lane] = 1.f;
    __builtin_amdgcn_wave_barrier();
    for (int sweep = 0; sweep < 8; ++sweep)
        for (int p = 0; p < ATT_N - 1; ++p)
            for (int q = p + 1; q < ATT_N; ++q) {
                float apq = A[p * JLD + q];
                float app = A[p * JLD + p];
                float aqq = A[q * JLD + q];
                float c = 1.f, s = 0.f;
                if (fabsf(apq) > 1e-20f) {
                    float tau = (aqq - app) / (2.f * apq);
                    float tt  = 1.f / (fabsf(tau) + sqrtf(1.f + tau * tau));
                    if (tau < 0.f) tt = -tt;
                    c = rsqrtf(1.f + tt * tt);
                    s = tt * c;
                }
                float aip = 0.f, aiq = 0.f, vip = 0.f, viq = 0.f;
                if (lane < ATT_N) {
                    aip = A[lane * JLD + p]; aiq = A[lane * JLD + q];
                    vip = V[lane * JLD + p]; viq = V[lane * JLD + q];
                }
                __builtin_amdgcn_wave_barrier();
                if (lane < ATT_N) {
                    if (lane != p && lane != q) {
                        float np_ = c * aip - s * aiq;
                        float nq_ = s * aip + c * aiq;
                        A[lane * JLD + p] = np_; A[p * JLD + lane] = np_;
                        A[lane * JLD + q] = nq_; A[q * JLD + lane] = nq_;
                    } else if (lane == p) {
                        A[p * JLD + p] = c * c * app - 2.f * c * s * apq + s * s * aqq;
                        A[p * JLD + q] = 0.f;
                    } else {
                        A[q * JLD + q] = s * s * app + 2.f * c * s * apq + c * c * aqq;
                        A[q * JLD + p] = 0.f;
                    }
                    V[lane * JLD + p] = c * vip - s * viq;
                    V[lane * JLD + q] = s * vip + c * viq;
                }
                __builtin_amdgcn_wave_barrier();
            }
}

// =====================================================================
// K5: BiMap (W C W^T) + logm via eigh, one wave per matrix (Q/K/V x 8192)
// =====================================================================
#define W5 (400 + 360 + 360 + 342 + 342 + 24)   // 1828 floats/wave
__global__ void bimap_logm(const float* __restrict__ cov,
                           const float* __restrict__ wq, const float* __restrict__ wk,
                           const float* __restrict__ wv,
                           float* __restrict__ lq, float* __restrict__ lk, float* __restrict__ lv) {
    __shared__ float sw[8][W5];
    int wave = threadIdx.x >> 5, lane = threadIdx.x & 31;
    int job   = blockIdx.x * 8 + wave;          // < 24576
    int which = job / 8192;
    int bp    = job - which * 8192;
    float* cv  = sw[wave];
    float* Wm  = cv + 400;
    float* tmp = Wm + 360;
    float* A   = tmp + 360;
    float* V   = A + 342;
    float* fw  = V + 342;
    const float* wsrc = (which == 0) ? wq : ((which == 1) ? wk : wv);
    float* dst        = (which == 0) ? lq : ((which == 1) ? lk : lv);

    for (int e = lane; e < 400; e += 32) cv[e] = cov[(size_t)bp * 400 + e];
    for (int e = lane; e < 360; e += 32) Wm[e] = wsrc[e];
    __builtin_amdgcn_wave_barrier();
    for (int e = lane; e < 360; e += 32) {
        int i = e / C_TEMP, cc = e - i * C_TEMP;
        float acc = 0.f;
        for (int c2 = 0; c2 < C_TEMP; ++c2) acc += Wm[i * C_TEMP + c2] * cv[c2 * C_TEMP + cc];
        tmp[e] = acc;
    }
    __builtin_amdgcn_wave_barrier();
    for (int e = lane; e < TANG; e += 32) {
        int i, j; tri_decode(e, ATT_N, i, j);
        float acc = 0.f;
        for (int c = 0; c < C_TEMP; ++c) acc += tmp[i * C_TEMP + c] * Wm[j * C_TEMP + c];
        A[i * JLD + j] = acc; A[j * JLD + i] = acc;
    }
    __builtin_amdgcn_wave_barrier();
    jacobi18(A, V, lane);
    if (lane < ATT_N) fw[lane] = logf(fmaxf(A[lane * JLD + lane], 1e-10f));
    __builtin_amdgcn_wave_barrier();
    for (int e = lane; e < ATT_N * ATT_N; e += 32) {
        int i = e / ATT_N, j = e - i * ATT_N;
        float acc = 0.f;
        for (int k = 0; k < ATT_N; ++k) acc += V[i * JLD + k] * fw[k] * V[j * JLD + k];
        dst[(size_t)bp * 324 + e] = acc;
    }
}

// =====================================================================
// K6: log-Euclidean attention + fused expm->reeig->logm (= eigh of S
//     with eigenvalue map max(w, ln 1e-4)) + triu vectorization
// =====================================================================
__global__ void attn_eig(const float* __restrict__ lq, const float* __restrict__ lk,
                         const float* __restrict__ lv, float* __restrict__ feat) {
    __shared__ float Lq[8 * 324], Lk[8 * 324], Lv[8 * 324];
    __shared__ float sq[8], sk[8], prob[64];
    __shared__ float AV[8][342 + 342 + 24];
    int b = blockIdx.x, tid = threadIdx.x;
    for (int e = tid; e < 2592; e += 256) {
        Lq[e] = lq[(size_t)b * 2592 + e];
        Lk[e] = lk[(size_t)b * 2592 + e];
        Lv[e] = lv[(size_t)b * 2592 + e];
    }
    __syncthreads();
    if (tid < 8) {
        float s = 0.f;
        for (int e = 0; e < 324; ++e) { float v = Lq[tid * 324 + e]; s += v * v; }
        sq[tid] = s;
    } else if (tid < 16) {
        int j = tid - 8;
        float s = 0.f;
        for (int e = 0; e < 324; ++e) { float v = Lk[j * 324 + e]; s += v * v; }
        sk[j] = s;
    }
    __syncthreads();
    if (tid < 64) {
        int i = tid >> 3, j = tid & 7;
        float cr = 0.f;
        for (int e = 0; e < 324; ++e) cr += Lq[i * 324 + e] * Lk[j * 324 + e];
        float d2 = fmaxf(sq[i] + sk[j] - 2.f * cr, 0.f);
        float energy = sqrtf(d2 + 1e-12f);
        prob[i * 8 + j] = 1.f / (1.f + log1pf(energy));      // attention weights
    }
    __syncthreads();
    if (tid < 8) {                                           // softmax over queries i
        int j = tid;
        float mx = -1e30f;
        for (int i = 0; i < 8; ++i) mx = fmaxf(mx, prob[i * 8 + j]);
        float ex[8], sm = 0.f;
        for (int i = 0; i < 8; ++i) { ex[i] = expf(prob[i * 8 + j] - mx); sm += ex[i]; }
        float inv = 1.f / sm;
        for (int i = 0; i < 8; ++i) prob[i * 8 + j] = ex[i] * inv;
    }
    __syncthreads();

    int wave = tid >> 5, lane = tid & 31;
    int j = wave;                                            // output patch index
    float* A  = AV[wave];
    float* V  = A + 342;
    float* fw = A + 684;
    for (int e = lane; e < 324; e += 32) {
        float acc = 0.f;
        for (int i = 0; i < 8; ++i) acc += prob[i * 8 + j] * Lv[i * 324 + e];
        int r = e / ATT_N, cc = e - r * ATT_N;
        A[r * JLD + cc] = acc;                               // S_j (symmetric)
    }
    __builtin_amdgcn_wave_barrier();
    jacobi18(A, V, lane);
    if (lane < ATT_N)
        fw[lane] = fmaxf(A[lane * JLD + lane], -9.210340371976184f);  // ln(1e-4)
    __builtin_amdgcn_wave_barrier();
    for (int e = lane; e < TANG; e += 32) {
        int i, jj; tri_decode(e, ATT_N, i, jj);
        float acc = 0.f;
        for (int k = 0; k < ATT_N; ++k) acc += V[i * JLD + k] * fw[k] * V[jj * JLD + k];
        float sc = (i == jj) ? 1.f : 1.41421356237309505f;
        feat[(size_t)b * FEATD + j * TANG + e] = acc * sc;
    }
}

// =====================================================================
// K7: final linear (B,1368) x (1368,4)
// =====================================================================
__global__ void final_lin(const float* __restrict__ feat, const float* __restrict__ w_lin,
                          const float* __restrict__ b_lin, float* __restrict__ out) {
    __shared__ float red[128 * N_OUT];
    int b = blockIdx.x, tid = threadIdx.x;
    float acc[N_OUT] = {0.f, 0.f, 0.f, 0.f};
    for (int d = tid; d < FEATD; d += 128) {
        float f = feat[(size_t)b * FEATD + d];
        #pragma unroll
        for (int n = 0; n < N_OUT; ++n) acc[n] += f * w_lin[n * FEATD + d];
    }
    #pragma unroll
    for (int n = 0; n < N_OUT; ++n) red[n * 128 + tid] = acc[n];
    __syncthreads();
    for (int s = 64; s > 0; s >>= 1) {
        if (tid < s)
            #pragma unroll
            for (int n = 0; n < N_OUT; ++n) red[n * 128 + tid] += red[n * 128 + tid + s];
        __syncthreads();
    }
    if (tid < N_OUT) out[b * N_OUT + tid] = red[tid * 128] + b_lin[tid];
}

// =====================================================================
extern "C" void kernel_launch(void* const* d_in, const int* in_sizes, int n_in,
                              void* d_out, int out_size, void* d_ws, size_t ws_size,
                              hipStream_t stream) {
    (void)in_sizes; (void)n_in; (void)out_size; (void)ws_size;
    const float* x      = (const float*)d_in[0];
    const float* w_spat = (const float*)d_in[1];
    const float* b_spat = (const float*)d_in[2];
    const float* g1     = (const float*)d_in[3];
    const float* be1    = (const float*)d_in[4];
    const float* w_temp = (const float*)d_in[5];
    /* b_temp (d_in[6]) cancels under patch centering */
    const float* g2     = (const float*)d_in[7];
    /* be2 (d_in[8]) cancels under patch centering */
    const float* wq     = (const float*)d_in[9];
    const float* wk     = (const float*)d_in[10];
    const float* wv     = (const float*)d_in[11];
    const float* w_lin  = (const float*)d_in[12];
    const float* b_lin  = (const float*)d_in[13];

    char* ws = (char*)d_ws;
    float*    st   = (float*)(ws + OFF_STAT);
    _Float16* Aw   = (_Float16*)(ws + OFF_AW);
    _Float16* h1   = (_Float16*)(ws + OFF_H1);
    float*    y    = (float*)(ws + OFF_Y);
    float*    cov  = (float*)(ws + OFF_COV);
    float*    lq   = (float*)(ws + OFF_LQ);
    float*    lk   = (float*)(ws + OFF_LK);
    float*    lv   = (float*)(ws + OFF_LV);
    float*    feat = (float*)(ws + OFF_FEAT);
    float*    out  = (float*)d_out;

    (void)hipMemsetAsync(st, 0, 4096, stream);
    prep_weights<<<36, 256, 0, stream>>>(w_temp, Aw);
    gram_x<<<BATCH * 8, 256, 0, stream>>>(x, st);
    bn1_finalize<<<1, 32, 0, stream>>>(w_spat, b_spat, g1, be1, st);
    spatial_bn<<<BATCH * 8, 256, 0, stream>>>(x, w_spat, st, h1);
    conv_wmma<<<BATCH * 16, 256, 0, stream>>>(Aw, h1, y);
    bn2_stats<<<BATCH * C_TEMP, 256, 0, stream>>>(y, st);
    bn2_finalize<<<1, 32, 0, stream>>>(g2, st);
    cov_k<<<BATCH * N_PATCH, 256, 0, stream>>>(y, st, cov);
    bimap_logm<<<3072, 256, 0, stream>>>(cov, wq, wk, wv, lq, lk, lv);
    attn_eig<<<BATCH, 256, 0, stream>>>(lq, lk, lv, feat);
    final_lin<<<BATCH, 128, 0, stream>>>(feat, w_lin, b_lin, out);
}
